// AutoregressiveDecoder_32315333935121
// MI455X (gfx1250) — compile-verified, matrix operand order checked by reading, not run on hardware
//
#include <hip/hip_runtime.h>

// ---------------------------------------------------------------------------
// Persistent GRU decoder for MI455X (gfx1250, wave32, WMMA bf16->f32).
// B=4096, H=128, LAT=32, IN=3, STEPS=512.
//
// Grid: 64 blocks x 256 threads (8 waves). Each wave-pair owns a 16-row
// batch tile; even wave handles column-tiles 0..3, odd wave 4..7.
// Weights pre-swizzled into WMMA B-fragment order in LDS (once).
// z-dependent gate contributions precomputed once into C-fragment order.
// ---------------------------------------------------------------------------

typedef __attribute__((ext_vector_type(16))) __bf16 v16bf;
typedef __attribute__((ext_vector_type(8)))  __bf16 v8bf;
typedef __attribute__((ext_vector_type(8)))  float  v8f;

#define STEPS   512
#define HDIM    128
#define LATD    32
#define ROWS_WG 64     // batch rows per workgroup (4 M-tiles)
#define HB_PITCH 136   // bf16 h row pitch (16B-aligned, bank-skewed)

// LDS region sizes (bytes)
#define WF_ELEMS  (124 * 32 * 16)          // 120 gate frags + 4 W_out frags
#define WF_BYTES  (WF_ELEMS * 2)           // 126976
#define GZ_ELEMS  (4 * 24 * 32 * 8)        // C-frag order f32
#define GZ_BYTES  (GZ_ELEMS * 4)           // 98304
#define H32_ELEMS (4 * 8 * 32 * 8)         // fp32 h state, C-frag order
#define H32_BYTES (H32_ELEMS * 4)          // 32768
#define HB_BYTES  (ROWS_WG * HB_PITCH * 2) // 17408
#define POS_ELEMS (ROWS_WG * 4)
#define POS_BYTES (POS_ELEMS * 2)          // 512
#define BHH_BYTES (384 * 4)
#define BOUT_BYTES 16
#define SMEM_TOTAL (WF_BYTES + GZ_BYTES + H32_BYTES + HB_BYTES + POS_BYTES + BHH_BYTES + BOUT_BYTES)

static __device__ __forceinline__ v8f wmma_bf16(v16bf a, v16bf b, v8f c) {
  return __builtin_amdgcn_wmma_f32_16x16x32_bf16(false, a, false, b, (short)0, c,
                                                 false, false);
}

static __device__ __forceinline__ float fast_sigmoid(float x) {
  return 1.0f / (1.0f + __expf(-x));
}

__global__ void __launch_bounds__(256, 1)
gru_decoder_kernel(const float* __restrict__ ctx,   // [4096,128]
                   const float* __restrict__ zin,   // [4096,32]
                   const float* __restrict__ Wih,   // [384,35]
                   const float* __restrict__ Whh,   // [384,128]
                   const float* __restrict__ bih,   // [384]
                   const float* __restrict__ bhh,   // [384]
                   const float* __restrict__ Wout,  // [3,128]
                   const float* __restrict__ bout,  // [3]
                   float* __restrict__ out)         // [4096,512,3]
{
  extern __shared__ char smem[];
  __bf16* wfrag = (__bf16*)smem;                                   // [124][32][16]
  float*  gzf   = (float*)(smem + WF_BYTES);                       // [4][24][32][8]
  float*  h32f  = (float*)(smem + WF_BYTES + GZ_BYTES);            // [4][8][32][8]
  __bf16* hB    = (__bf16*)(smem + WF_BYTES + GZ_BYTES + H32_BYTES);
  __bf16* posL  = (__bf16*)((char*)hB + HB_BYTES);                 // [64][4]
  float*  bhhL  = (float*)((char*)posL + POS_BYTES);               // [384]
  float*  boutL = (float*)((char*)bhhL + BHH_BYTES);               // [4]

  const int tid      = threadIdx.x;
  const int lane     = tid & 31;
  const int wave     = tid >> 5;
  const int pr       = wave >> 1;   // M-tile index 0..3
  const int chalf    = wave & 1;    // column-tile half
  const int nl       = lane & 15;
  const int hl       = lane >> 4;
  const int blockRow = blockIdx.x * ROWS_WG;
  const int p16      = pr * 16;

  // ---------------- one-time precompute --------------------------------
  // (1) weights -> WMMA B-fragment layout (bf16)
  for (int i = tid; i < WF_ELEMS; i += 256) {
    int f = i >> 9;
    int rem = i & 511;
    int ln = rem >> 4;
    int e  = rem & 15;
    int nn = ln & 15, hh = ln >> 4;
    int kk = ((e >> 3) << 4) + (hh << 3) + (e & 7);  // K within 32-block
    float v;
    if (f < 120) {
      int t = f / 5, kt = f % 5;
      int col = t * 16 + nn;                 // output column in 3H
      if (kt < 4) v = Whh[col * HDIM + kt * 32 + kk];
      else        v = (kk < 3) ? Wih[col * 35 + kk] : 0.0f;  // pos weights
    } else {
      int kt = f - 120;
      v = (nn < 3) ? Wout[nn * HDIM + kt * 32 + kk] : 0.0f;
    }
    wfrag[i] = (__bf16)v;
  }
  // (2) z-path gate preactivations -> C-fragment order (f32)
  for (int i = tid; i < GZ_ELEMS; i += 256) {
    int r  = i & 7;
    int ln = (i >> 3) & 31;
    int t  = (i >> 8) % 24;
    int mt = i / 6144;
    int rowg = blockRow + mt * 16 + r + ((ln >> 4) << 3);
    int col  = t * 16 + (ln & 15);
    float acc = bih[col] + (col < 256 ? bhh[col] : 0.0f);
    const float* zr = zin + rowg * LATD;
    const float* wz = Wih + col * 35 + 3;
#pragma unroll
    for (int j = 0; j < LATD; ++j) acc = fmaf(zr[j], wz[j], acc);
    gzf[i] = acc;
  }
  // (3) h state: fp32 C-frag order + bf16 row-major (A operand source)
  for (int i = tid; i < H32_ELEMS; i += 256) {
    int r  = i & 7;
    int ln = (i >> 3) & 31;
    int ct = (i >> 8) & 7;
    int mt = i >> 11;
    int rowg = blockRow + mt * 16 + r + ((ln >> 4) << 3);
    int col  = ct * 16 + (ln & 15);
    h32f[i] = ctx[rowg * HDIM + col];
  }
  for (int i = tid; i < ROWS_WG * HDIM; i += 256) {
    int rl = i >> 7, c = i & 127;
    hB[rl * HB_PITCH + c] = (__bf16)ctx[(blockRow + rl) * HDIM + c];
  }
  for (int i = tid; i < POS_ELEMS; i += 256) posL[i] = (__bf16)0.0f;
  for (int i = tid; i < 384; i += 256) bhhL[i] = bhh[i];
  if (tid < 3) boutL[tid] = bout[tid];
  __syncthreads();

  // ---------------- helpers --------------------------------------------
  auto loadW = [&](int f) -> v16bf {
    const __bf16* wp = wfrag + (((unsigned)f << 5) + lane) * 16;
    v8bf lo = *(const v8bf*)wp;
    v8bf hi = *(const v8bf*)(wp + 8);
    return __builtin_shufflevector(lo, hi, 0, 1, 2, 3, 4, 5, 6, 7, 8, 9, 10, 11,
                                   12, 13, 14, 15);
  };
  auto loadA = [&](int kt) -> v16bf {
    // 16-bit A 16x32 layout: lane holds rows m=lane&15; two contiguous
    // 8-element K runs at 32*kt + 8*hl and 32*kt + 16 + 8*hl.
    const __bf16* ap = hB + (p16 + nl) * HB_PITCH + kt * 32 + hl * 8;
    v8bf lo = *(const v8bf*)ap;
    v8bf hi = *(const v8bf*)(ap + 16);
    return __builtin_shufflevector(lo, hi, 0, 1, 2, 3, 4, 5, 6, 7, 8, 9, 10, 11,
                                   12, 13, 14, 15);
  };

  // ---------------- recurrent main loop --------------------------------
#pragma unroll 1
  for (int s = 0; s < STEPS; ++s) {
    // A fragments of current h (K = 0..127)
    v16bf A0 = loadA(0), A1 = loadA(1), A2 = loadA(2), A3 = loadA(3);
    // pos A fragment (K tile 4: K=0..2 valid, rest zero)
    v16bf Ap;
#pragma unroll
    for (int e = 0; e < 16; ++e) Ap[e] = (__bf16)0.0f;
    if (hl == 0) {
      const __bf16* pp = posL + (p16 + nl) * 4;
      Ap[0] = pp[0]; Ap[1] = pp[1]; Ap[2] = pp[2];
    }

#pragma unroll 1
    for (int i2 = 0; i2 < 4; ++i2) {
      const int ct = chalf * 4 + i2;
      // C init: precomputed z-path + biases (r,u merged; n split)
      v8f accR = *(const v8f*)(gzf + (((pr * 24) + ct) * 32 + lane) * 8);
      v8f accZ = *(const v8f*)(gzf + (((pr * 24) + 8 + ct) * 32 + lane) * 8);
      v8f accN = *(const v8f*)(gzf + (((pr * 24) + 16 + ct) * 32 + lane) * 8);
      float bh = bhhL[256 + ct * 16 + nl];
      v8f accH;
#pragma unroll
      for (int j = 0; j < 8; ++j) accH[j] = bh;

      accR = wmma_bf16(A0, loadW(ct * 5 + 0), accR);
      accZ = wmma_bf16(A0, loadW((8 + ct) * 5 + 0), accZ);
      accH = wmma_bf16(A0, loadW((16 + ct) * 5 + 0), accH);
      accR = wmma_bf16(A1, loadW(ct * 5 + 1), accR);
      accZ = wmma_bf16(A1, loadW((8 + ct) * 5 + 1), accZ);
      accH = wmma_bf16(A1, loadW((16 + ct) * 5 + 1), accH);
      accR = wmma_bf16(A2, loadW(ct * 5 + 2), accR);
      accZ = wmma_bf16(A2, loadW((8 + ct) * 5 + 2), accZ);
      accH = wmma_bf16(A2, loadW((16 + ct) * 5 + 2), accH);
      accR = wmma_bf16(A3, loadW(ct * 5 + 3), accR);
      accZ = wmma_bf16(A3, loadW((8 + ct) * 5 + 3), accZ);
      accH = wmma_bf16(A3, loadW((16 + ct) * 5 + 3), accH);
      accR = wmma_bf16(Ap, loadW(ct * 5 + 4), accR);
      accZ = wmma_bf16(Ap, loadW((8 + ct) * 5 + 4), accZ);
      accN = wmma_bf16(Ap, loadW((16 + ct) * 5 + 4), accN);

      // GRU gate epilogue; fp32 state in LDS, bf16 mirror for next A
      float* hp = h32f + (((pr * 8) + ct) * 32 + lane) * 8;
      v8f hold = *(const v8f*)hp;
      v8f hnew;
#pragma unroll
      for (int j = 0; j < 8; ++j) {
        float rr = fast_sigmoid(accR[j]);
        float uu = fast_sigmoid(accZ[j]);
        float nn = tanhf(accN[j] + rr * accH[j]);
        hnew[j] = nn + uu * (hold[j] - nn);
      }
      *(v8f*)hp = hnew;
#pragma unroll
      for (int j = 0; j < 8; ++j)
        hB[(p16 + j + hl * 8) * HB_PITCH + ct * 16 + nl] = (__bf16)hnew[j];
    }

    __syncthreads();  // full h_new visible to the pair

    if (chalf == 0) {
      // next_pos = h_new @ W_out^T + b_out  (N=3 packed in a 16-wide tile)
      v16bf B0 = loadA(0), B1 = loadA(1), B2 = loadA(2), B3 = loadA(3);
      float pb = (nl < 3) ? boutL[nl] : 0.0f;
      v8f pc;
#pragma unroll
      for (int j = 0; j < 8; ++j) pc[j] = pb;
      pc = wmma_bf16(B0, loadW(120), pc);
      pc = wmma_bf16(B1, loadW(121), pc);
      pc = wmma_bf16(B2, loadW(122), pc);
      pc = wmma_bf16(B3, loadW(123), pc);
      if (nl < 3) {
#pragma unroll
        for (int j = 0; j < 8; ++j) {
          int rloc = p16 + j + hl * 8;
          int row  = blockRow + rloc;
          out[(size_t)row * (STEPS * 3) + s * 3 + nl] = pc[j];
          posL[rloc * 4 + nl] = (__bf16)pc[j];
        }
      }
    }
    __syncthreads();  // pos visible before next step's A_pos load
  }
}

extern "C" void kernel_launch(void* const* d_in, const int* in_sizes, int n_in,
                              void* d_out, int out_size, void* d_ws, size_t ws_size,
                              hipStream_t stream) {
  (void)in_sizes; (void)n_in; (void)out_size; (void)d_ws; (void)ws_size;
  const float* ctx  = (const float*)d_in[0];
  const float* zin  = (const float*)d_in[1];
  // d_in[2] = steps (int, ==512; compile-time constant in kernel)
  const float* Wih  = (const float*)d_in[3];
  const float* Whh  = (const float*)d_in[4];
  const float* bih  = (const float*)d_in[5];
  const float* bhh  = (const float*)d_in[6];
  const float* Wout = (const float*)d_in[7];
  const float* bout = (const float*)d_in[8];
  float* out = (float*)d_out;

  hipFuncSetAttribute((const void*)gru_decoder_kernel,
                      hipFuncAttributeMaxDynamicSharedMemorySize, SMEM_TOTAL);

  dim3 grid(4096 / ROWS_WG);  // 64
  dim3 block(256);
  gru_decoder_kernel<<<grid, block, SMEM_TOTAL, stream>>>(
      ctx, zin, Wih, Whh, bih, bhh, Wout, bout, out);
}
